// AsymmetricRevVit_39702677684399
// MI455X (gfx1250) — compile-verified, matrix operand order checked by reading, not run on hardware
//
#include <hip/hip_runtime.h>
#include <hip/hip_bf16.h>

#define NB 8
#define NHD 8
#define HD 96
#define DC 768
#define N3 2304
#define HQ 56
#define NQTOK 3137      // 56*56+1
#define HK 28
#define NKTOK 785       // 28*28+1
#define BHN (NB*NHD)    // 64
#define SPQ 3136
#define QTILES 197      // ceil(3137/16)
#define VSTRIDE 800     // padded token stride for transposed V

typedef __attribute__((ext_vector_type(16))) __bf16 v16bf;
typedef __attribute__((ext_vector_type(8)))  __bf16 v8bf;
typedef __attribute__((ext_vector_type(8)))  float  v8f;

__device__ __forceinline__ __bf16 f2bf(float f){
  unsigned u = __builtin_bit_cast(unsigned, f);
  unsigned r = u + 0x7FFFu + ((u >> 16) & 1u);
  unsigned short s = (unsigned short)(r >> 16);
  return __builtin_bit_cast(__bf16, s);
}

// Per-lane fragment data = two contiguous 8-elem runs: [8h..8h+7] and [16+8h..+7].
// rp = lane's row pointer (row*stride + K-chunk offset applied). 16B-aligned.
__device__ __forceinline__ v16bf load_frag(const __bf16* rp, int h){
  v8bf lo = *(const v8bf*)(rp + 8*h);
  v8bf hi = *(const v8bf*)(rp + 16 + 8*h);
  return __builtin_shufflevector(lo, hi, 0,1,2,3,4,5,6,7,8,9,10,11,12,13,14,15);
}

__device__ __forceinline__ v8f wmma_bf16(v16bf a, v16bf b, v8f c){
  return __builtin_amdgcn_wmma_f32_16x16x32_bf16(false, a, false, b, (short)0, c,
                                                 false, false);
}

__device__ __forceinline__ float redmax16(float v){
#pragma unroll
  for (int m = 1; m < 16; m <<= 1) v = fmaxf(v, __shfl_xor(v, m, 32));
  return v;
}
__device__ __forceinline__ float redsum16(float v){
#pragma unroll
  for (int m = 1; m < 16; m <<= 1) v += __shfl_xor(v, m, 32);
  return v;
}
__device__ __forceinline__ float redsum32(float v){
#pragma unroll
  for (int m = 1; m < 32; m <<= 1) v += __shfl_xor(v, m, 32);
  return v;
}

// ---------------- f32 -> bf16, 8 elements/thread (float4 x2 in, 16B out) ------
__global__ __launch_bounds__(256) void cvt8_kernel(const float* __restrict__ in,
                                                   __bf16* __restrict__ out,
                                                   long n8){
  long i = (long)blockIdx.x * 256 + threadIdx.x;
  if (i >= n8) return;
  const float4* p = (const float4*)in + i*2;
  float4 a = p[0], b = p[1];
  v8bf o;
  o[0]=f2bf(a.x); o[1]=f2bf(a.y); o[2]=f2bf(a.z); o[3]=f2bf(a.w);
  o[4]=f2bf(b.x); o[5]=f2bf(b.y); o[6]=f2bf(b.z); o[7]=f2bf(b.w);
  *((v8bf*)out + i) = o;
}

// ---------------- qkv_w [768][2304] f32 -> wT [2304][768] bf16 (LDS tiled) ----
__global__ __launch_bounds__(256) void cvt_transpose_kernel(const float* __restrict__ in,
                                                            __bf16* __restrict__ out){
  __shared__ float tile[32][33];
  const int j0 = blockIdx.x * 32;   // output row block (N3 dim)
  const int c0 = blockIdx.y * 32;   // K dim block
  const int tx = threadIdx.x & 31;
  const int ty = threadIdx.x >> 5;  // 0..7
#pragma unroll
  for (int i = ty; i < 32; i += 8)
    tile[i][tx] = in[(size_t)(c0 + i)*N3 + j0 + tx];
  __syncthreads();
#pragma unroll
  for (int i = ty; i < 32; i += 8)
    out[(size_t)(j0 + i)*DC + c0 + tx] = f2bf(tile[tx][i]);
}

// ---------------- QKV GEMM (bf16 WMMA, f32 acc) -------------------------------
// xb: [8][3137][768] bf16, wT: [2304][768] bf16 (pre-transposed).
// grid = (18, 197, 8); block = 128 (4 waves, each a 16x32 tile).
__global__ __launch_bounds__(128) void qkv_gemm_kernel(const __bf16* __restrict__ xb,
                                                       const __bf16* __restrict__ wT,
                                                       const float*  __restrict__ bias,
                                                       float* __restrict__ qo,
                                                       float* __restrict__ ko,
                                                       float* __restrict__ vo){
  const int wave = threadIdx.x >> 5;
  const int lane = threadIdx.x & 31;
  const int col  = lane & 15;
  const int half = lane >> 4;
  const int b_ = blockIdx.z;
  const int m0 = blockIdx.y * 16;                 // token tile within batch
  const int n0 = blockIdx.x * 128 + wave * 32;
  const int rr = min(m0 + col, NQTOK - 1);

  const __bf16* arow = xb + ((size_t)b_*NQTOK + rr)*DC;
  const int j0 = n0 + col, j1 = j0 + 16;
  const __bf16* brow0 = wT + (size_t)j0*DC;
  const __bf16* brow1 = wT + (size_t)j1*DC;

  v8f acc0 = {}, acc1 = {};
  for (int kc = 0; kc < DC; kc += 32){
    v16bf a  = load_frag(arow  + kc, half);
    v16bf b0 = load_frag(brow0 + kc, half);
    v16bf b1 = load_frag(brow1 + kc, half);
    acc0 = wmma_bf16(a, b0, acc0);
    acc1 = wmma_bf16(a, b1, acc1);
  }

#pragma unroll
  for (int f = 0; f < 2; ++f){
    v8f acc = f ? acc1 : acc0;
    int j = n0 + f*16 + col;
    float bv = bias[j];
    int which = j / 768;
    int rem   = j - which*768;
    int head  = rem / HD;
    int ch    = rem - head*HD;
    float* dst = (which == 0) ? qo : ((which == 1) ? ko : vo);
    size_t dbase = ((size_t)b_*NHD + head)*NQTOK;
#pragma unroll
    for (int r = 0; r < 8; ++r){
      int n_ = m0 + r + 8*half;
      if (n_ < NQTOK)
        dst[(dbase + n_)*HD + ch] = acc[r] + bv;
    }
  }
}

// ---------------- depthwise 3x3 pool + LayerNorm ------------------------------
// One wave per output token; lane owns 3 channels.
// tmode 0: outb normal [bh][NO][96]. tmode 1: outb transposed [bh][96][tstride],
// with zero fill for pad tokens (NO <= token < NOpad).
__global__ __launch_bounds__(256) void pool_ln_kernel(const float* __restrict__ in,
                                                      const float* __restrict__ w,
                                                      const float* __restrict__ g,
                                                      const float* __restrict__ beta,
                                                      float*  __restrict__ outf,
                                                      __bf16* __restrict__ outb,
                                                      int stride, int HO, int WO,
                                                      int tmode, int tstride,
                                                      int NOpad){
  const int wave = threadIdx.x >> 5;
  const int lane = threadIdx.x & 31;
  const int NO = HO*WO + 1;
  const int token = blockIdx.x * 8 + wave;
  const int bh = blockIdx.y;
  if (token >= NOpad) return;
  if (token >= NO){            // zero-fill transposed padding
    if (tmode){
      __bf16 z = f2bf(0.f);
#pragma unroll
      for (int i = 0; i < 3; ++i)
        outb[((size_t)bh*HD + lane*3 + i)*tstride + token] = z;
    }
    return;
  }
  const float* src = in + (size_t)bh * NQTOK * HD;

  float vals[3];
#pragma unroll
  for (int i = 0; i < 3; ++i){
    int c = lane*3 + i;
    float acc;
    if (token == 0){
      acc = src[c];                       // cls token: conv bypass
    } else {
      int sp = token - 1;
      int oy = sp / WO, ox = sp - oy*WO;
      acc = 0.f;
#pragma unroll
      for (int dy = 0; dy < 3; ++dy)
#pragma unroll
        for (int dx = 0; dx < 3; ++dx){
          int iy = oy*stride + dy - 1, ix = ox*stride + dx - 1;
          if (iy >= 0 && iy < HQ && ix >= 0 && ix < HQ)
            acc = fmaf(src[(size_t)(1 + iy*HQ + ix)*HD + c], w[(dy*3+dx)*HD + c], acc);
        }
    }
    vals[i] = acc;
  }
  float s = redsum32(vals[0] + vals[1] + vals[2]);
  float mean = s * (1.f/HD);
  float d0 = vals[0]-mean, d1 = vals[1]-mean, d2 = vals[2]-mean;
  float s2 = redsum32(d0*d0 + d1*d1 + d2*d2);
  float inv = rsqrtf(s2 * (1.f/HD) + 1e-5f);
#pragma unroll
  for (int i = 0; i < 3; ++i){
    int c = lane*3 + i;
    float o = (vals[i]-mean)*inv*g[c] + beta[c];
    if (outf) outf[((size_t)bh*NO + token)*HD + c] = o;
    if (tmode) outb[((size_t)bh*HD + c)*tstride + token] = f2bf(o);
    else       outb[((size_t)bh*NO + token)*HD + c]      = f2bf(o);
  }
}

// ---------------- rel-pos tables (float4 dot products) ------------------------
__global__ __launch_bounds__(256) void rel_kernel(const float* __restrict__ qpf,
                                                  const float* __restrict__ rph,
                                                  const float* __restrict__ rpw,
                                                  float* __restrict__ relh,
                                                  float* __restrict__ relw){
  long tid = (long)blockIdx.x * 256 + threadIdx.x;
  long total = (long)BHN * SPQ * HK;
  if (tid >= total) return;
  int kk = (int)(tid % HK);
  long rest = tid / HK;
  int sp = (int)(rest % SPQ);
  int bh = (int)(rest / SPQ);
  int y = sp / HQ, x = sp - y*HQ;
  const float4* q4 = (const float4*)(qpf + ((size_t)bh*NQTOK + 1 + sp)*HD);
  const float4* h4 = (const float4*)(rph + (size_t)(y - 2*kk + 54)*HD);
  const float4* w4 = (const float4*)(rpw + (size_t)(x - 2*kk + 54)*HD);
  float sh = 0.f, sw = 0.f;
#pragma unroll
  for (int c = 0; c < HD/4; ++c){
    float4 qv = q4[c], hv = h4[c], wv = w4[c];
    sh = fmaf(qv.x, hv.x, sh); sh = fmaf(qv.y, hv.y, sh);
    sh = fmaf(qv.z, hv.z, sh); sh = fmaf(qv.w, hv.w, sh);
    sw = fmaf(qv.x, wv.x, sw); sw = fmaf(qv.y, wv.y, sw);
    sw = fmaf(qv.z, wv.z, sw); sw = fmaf(qv.w, wv.w, sw);
  }
  size_t o = ((size_t)bh*SPQ + sp)*HK + kk;
  relh[o] = sh;
  relw[o] = sw;
}

// ---------------- flash attention (WMMA bf16, online softmax) -----------------
// One wave = 16 q-rows for one (b,head). K/V tiled by 32 tokens.
// vT: transposed pooled V [bh][96][VSTRIDE] bf16, zero-padded tokens.
__global__ __launch_bounds__(32) void attn_kernel(const __bf16* __restrict__ qb,
                                                  const float*  __restrict__ qf,
                                                  const __bf16* __restrict__ kb,
                                                  const __bf16* __restrict__ vT,
                                                  const float*  __restrict__ relh,
                                                  const float*  __restrict__ relw,
                                                  float* __restrict__ out){
  __shared__ __attribute__((aligned(16))) __bf16 lds_p[16*32];
  const int bh = blockIdx.y;
  const int q0 = blockIdx.x * 16;
  const int lane = threadIdx.x;
  const int col  = lane & 15;
  const int half = lane >> 4;
  const float scale = 0.1020620726159658f;   // 96^-0.5

  const __bf16* qB = qb + (size_t)bh*NQTOK*HD;
  const float*  qF = qf + (size_t)bh*NQTOK*HD;
  const __bf16* kB = kb + (size_t)bh*NKTOK*HD;
  const __bf16* vBase = vT + (size_t)bh*HD*VSTRIDE;
  const float*  rH = relh + (size_t)bh*SPQ*HK;
  const float*  rW = relw + (size_t)bh*SPQ*HK;

  // Q fragments: 3 chunks of K=32 over the 96-dim head
  v16bf qa[3];
  {
    const __bf16* qrow = qB + (size_t)min(q0 + col, NQTOK - 1)*HD;
#pragma unroll
    for (int c = 0; c < 3; ++c) qa[c] = load_frag(qrow + c*32, half);
  }

  v8f o[6];
#pragma unroll
  for (int j = 0; j < 6; ++j) o[j] = v8f{};
  float mrow[8], lsum[8];
#pragma unroll
  for (int r = 0; r < 8; ++r){ mrow[r] = -1e30f; lsum[r] = 0.f; }

  for (int c0 = 0; c0 < NKTOK; c0 += 32){
    // S = q . k^T  (k rows contiguous per lane)
    v8f s0 = {}, s1 = {};
    const __bf16* krow0 = kB + (size_t)min(c0 + col,      NKTOK - 1)*HD;
    const __bf16* krow1 = kB + (size_t)min(c0 + 16 + col, NKTOK - 1)*HD;
#pragma unroll
    for (int c = 0; c < 3; ++c){
      s0 = wmma_bf16(qa[c], load_frag(krow0 + c*32, half), s0);
      s1 = wmma_bf16(qa[c], load_frag(krow1 + c*32, half), s1);
    }

    // scale + rel-pos + mask, then online softmax update
    float p0[8], p1[8];
#pragma unroll
    for (int r = 0; r < 8; ++r){
      int tq = q0 + r + 8*half;
      int tk0 = c0 + col, tk1 = tk0 + 16;
      float v0 = s0[r]*scale, v1 = s1[r]*scale;
      if (tq >= 1 && tq < NQTOK){
        int sp = tq - 1;
        if (tk0 >= 1 && tk0 < NKTOK){
          int ks = tk0 - 1; int ky = ks / HK, kx = ks - ky*HK;
          v0 += rH[(size_t)sp*HK + ky] + rW[(size_t)sp*HK + kx];
        }
        if (tk1 >= 1 && tk1 < NKTOK){
          int ks = tk1 - 1; int ky = ks / HK, kx = ks - ky*HK;
          v1 += rH[(size_t)sp*HK + ky] + rW[(size_t)sp*HK + kx];
        }
      }
      if (tk0 >= NKTOK) v0 = -1e30f;
      if (tk1 >= NKTOK) v1 = -1e30f;

      float tm = redmax16(fmaxf(v0, v1));
      float nm = fmaxf(mrow[r], tm);
      float corr = __expf(mrow[r] - nm);
      mrow[r] = nm;
      lsum[r] *= corr;
      o[0][r] *= corr; o[1][r] *= corr; o[2][r] *= corr;
      o[3][r] *= corr; o[4][r] *= corr; o[5][r] *= corr;
      p0[r] = __expf(v0 - nm);
      p1[r] = __expf(v1 - nm);
      lsum[r] += p0[r] + p1[r];
    }

    // C-frag -> A-frag re-layout for P via LDS
#pragma unroll
    for (int r = 0; r < 8; ++r){
      int lrow = r + 8*half;
      lds_p[lrow*32 + col]      = f2bf(p0[r]);
      lds_p[lrow*32 + 16 + col] = f2bf(p1[r]);
    }
    __syncthreads();
    v16bf pa = load_frag(lds_p + col*32, half);
    __syncthreads();

    // O += P . V  (vT rows contiguous along tokens, zero-padded)
#pragma unroll
    for (int j = 0; j < 6; ++j){
      const __bf16* vrow = vBase + (size_t)(j*16 + col)*VSTRIDE;
      o[j] = wmma_bf16(pa, load_frag(vrow + c0, half), o[j]);
    }
  }

  // normalize, residual, transposed store
  const int b_ = bh >> 3, h_ = bh & 7;
#pragma unroll
  for (int r = 0; r < 8; ++r){
    float l = redsum16(lsum[r]);
    float inv = (l > 0.f) ? (1.f / l) : 0.f;
    int tq = q0 + r + 8*half;
    if (tq < NQTOK){
#pragma unroll
      for (int j = 0; j < 6; ++j){
        int ch = j*16 + col;
        float val = o[j][r]*inv + qF[(size_t)tq*HD + ch];
        out[((size_t)b_*NQTOK + tq)*DC + h_*HD + ch] = val;
      }
    }
  }
}

extern "C" void kernel_launch(void* const* d_in, const int* in_sizes, int n_in,
                              void* d_out, int out_size, void* d_ws, size_t ws_size,
                              hipStream_t stream) {
  const float* x      = (const float*)d_in[0];
  const float* qkv_w  = (const float*)d_in[1];
  const float* qkv_b  = (const float*)d_in[2];
  const float* pq_w   = (const float*)d_in[3];
  const float* pk_w   = (const float*)d_in[4];
  const float* pv_w   = (const float*)d_in[5];
  const float* nq_g   = (const float*)d_in[6];
  const float* nq_b   = (const float*)d_in[7];
  const float* nk_g   = (const float*)d_in[8];
  const float* nk_b   = (const float*)d_in[9];
  const float* nv_g   = (const float*)d_in[10];
  const float* nv_b   = (const float*)d_in[11];
  const float* rph    = (const float*)d_in[12];
  const float* rpw    = (const float*)d_in[13];
  float* out = (float*)d_out;

  char* base = (char*)d_ws;
  size_t cur = 0;
  auto alloc = [&](size_t bytes) -> void* {
    void* p = base + cur;
    cur += (bytes + 255) & ~(size_t)255;
    return p;
  };
  const size_t NXE = (size_t)NB * NQTOK * DC;   // 19,273,728
  const size_t NWE = (size_t)DC * N3;           // 1,769,472
  const size_t NQE = (size_t)BHN * NQTOK * HD;
  const size_t NKE = (size_t)BHN * NKTOK * HD;
  const size_t NVT = (size_t)BHN * HD * VSTRIDE;
  const size_t NRE = (size_t)BHN * SPQ * HK;

  __bf16* xb   = (__bf16*)alloc(NXE*2);
  __bf16* wT   = (__bf16*)alloc(NWE*2);
  float*  qpre = (float*) alloc(NQE*4);
  float*  kpre = (float*) alloc(NQE*4);
  float*  vpre = (float*) alloc(NQE*4);
  float*  qpf  = (float*) alloc(NQE*4);
  __bf16* qpb  = (__bf16*)alloc(NQE*2);
  __bf16* kpb  = (__bf16*)alloc(NKE*2);
  __bf16* vpt  = (__bf16*)alloc(NVT*2);
  float*  relh = (float*) alloc(NRE*4);
  float*  relw = (float*) alloc(NRE*4);

  // 1. bf16 conversions (x straight, w transposed)
  cvt8_kernel<<<(int)((NXE/8 + 255)/256), 256, 0, stream>>>(x, xb, (long)(NXE/8));
  cvt_transpose_kernel<<<dim3(N3/32, DC/32), 256, 0, stream>>>(qkv_w, wT);

  // 2. QKV projection GEMM (WMMA)
  qkv_gemm_kernel<<<dim3(N3/128, QTILES, NB), 128, 0, stream>>>(
      xb, wT, qkv_b, qpre, kpre, vpre);

  // 3. pool + LN  (q: normal layouts; k: normal bf16; v: transposed+padded bf16)
  pool_ln_kernel<<<dim3((NQTOK+7)/8, BHN), 256, 0, stream>>>(
      qpre, pq_w, nq_g, nq_b, qpf, qpb, 1, HQ, HQ, 0, 0, NQTOK);
  pool_ln_kernel<<<dim3((NKTOK+7)/8, BHN), 256, 0, stream>>>(
      kpre, pk_w, nk_g, nk_b, nullptr, kpb, 2, HK, HK, 0, 0, NKTOK);
  pool_ln_kernel<<<dim3((VSTRIDE+7)/8, BHN), 256, 0, stream>>>(
      vpre, pv_w, nv_g, nv_b, nullptr, vpt, 2, HK, HK, 1, VSTRIDE, VSTRIDE);

  // 4. rel-pos tables
  {
    long total = (long)BHN * SPQ * HK;
    rel_kernel<<<(int)((total+255)/256), 256, 0, stream>>>(qpf, rph, rpw, relh, relw);
  }

  // 5. flash attention + residual + transposed store
  attn_kernel<<<dim3(QTILES, BHN), 32, 0, stream>>>(
      qpb, qpf, kpb, vpt, relh, relw, out);

  (void)in_sizes; (void)n_in; (void)out_size; (void)ws_size;
}
